// GrassmanianEncoderLayer_44753559225154
// MI455X (gfx1250) — matrix operand order repaired and checked by reading, not compiled, not run on hardware
//
#include <hip/hip_runtime.h>
#include <hip/hip_bf16.h>

// ---------------------------------------------------------------------------
// Types
// ---------------------------------------------------------------------------
typedef __attribute__((ext_vector_type(16))) __bf16 v16bf;
typedef __attribute__((ext_vector_type(8)))  __bf16 v8bf;
typedef __attribute__((ext_vector_type(8)))  float  v8f;

union V16U { v16bf v; v8bf h[2]; };

static __device__ __forceinline__ __bf16 f2bf(float f) { return (__bf16)f; }

// ---------------------------------------------------------------------------
// Problem constants
// ---------------------------------------------------------------------------
#define BB   4
#define NN   1024
#define CC   768
#define HH   12
#define DD   64
#define DFF_ 3072
#define EPS_ 1e-5f

// ---------------------------------------------------------------------------
// Async global->LDS copy path (gfx1250). Guarded so the code still builds if
// the toolchain lacks the builtins; fallback is sync load + ds_store.
// Builtin signature (from hipcc diagnostic): param0 = <4 x i32> addrspace(1)*,
// param1 = LDS pointer, then i32 offset, i32 cpol.
// ---------------------------------------------------------------------------
#if defined(__HIP_DEVICE_COMPILE__) &&                                   \
    __has_builtin(__builtin_amdgcn_global_load_async_to_lds_b128) &&     \
    __has_builtin(__builtin_amdgcn_s_wait_asynccnt)
#define USE_ASYNC_LDS 1
#else
#define USE_ASYNC_LDS 0
#endif

#if USE_ASYNC_LDS
typedef int vi4_gcc __attribute__((vector_size(16)));
static __device__ __forceinline__ void async_cp16(const void* g, void* l) {
  __builtin_amdgcn_global_load_async_to_lds_b128(
      (__attribute__((address_space(1))) vi4_gcc*)g,
      (__attribute__((address_space(3))) vi4_gcc*)l, 0, 0);
}
#endif

// ---------------------------------------------------------------------------
// bf16 WMMA GEMM: D(M,N) = A(M,K) * B(N,K)^T   (B stored N-major, K-contig,
// matching every `x @ W.T` in the reference).
// Block tile 128x64, K-step 32, 256 threads = 8 waves; each wave computes a
// 16x64 strip (4 x v_wmma_f32_16x16x32_bf16 per K-step). Double-buffered LDS
// filled by GLOBAL_LOAD_ASYNC_TO_LDS_B128 (ASYNCcnt pipelined) when available.
// Batched over blockIdx.z.
// ---------------------------------------------------------------------------
template <class Epi>
__global__ __launch_bounds__(256) void gemm128(
    const __bf16* __restrict__ A, const __bf16* __restrict__ B,
    int K, int lda, int ldb, long long sAz, long long sBz, Epi epi) {
  __shared__ __bf16 As[2][128 * 32];
  __shared__ __bf16 Bs[2][64 * 32];

  const int tid  = threadIdx.x;
  const int wave = tid >> 5, lane = tid & 31;
  const int half = lane >> 4;   // lane group within wave
  const int mrow = lane & 15;
  const int z     = blockIdx.z;
  const int mBase = blockIdx.y * 128;
  const int nBase = blockIdx.x * 64;

  const __bf16* Ag = A + (size_t)z * sAz + (size_t)mBase * lda;
  const __bf16* Bg = B + (size_t)z * sBz + (size_t)nBase * ldb;

  v8f acc[4];
#pragma unroll
  for (int c = 0; c < 4; ++c) acc[c] = (v8f){0, 0, 0, 0, 0, 0, 0, 0};

  const int lr = tid >> 2;        // 0..63 tile row/col loaded by this thread
  const int lk = (tid & 3) << 3;  // 0,8,16,24 : K offset (8 bf16 = 16B)
  const int steps = K >> 5;

#if USE_ASYNC_LDS
  // ---- software-pipelined double buffer via async Global->LDS DMA ----
  {  // preload tile 0
    async_cp16(Ag + (size_t)lr * lda + lk,        &As[0][lr * 32 + lk]);
    async_cp16(Ag + (size_t)(lr + 64) * lda + lk, &As[0][(lr + 64) * 32 + lk]);
    async_cp16(Bg + (size_t)lr * ldb + lk,        &Bs[0][lr * 32 + lk]);
  }
  for (int i = 0; i < steps; ++i) {
    const int cur = i & 1;
    if (i + 1 < steps) {  // issue tile i+1 into the other buffer
      const int ks = (i + 1) << 5;
      async_cp16(Ag + (size_t)lr * lda + ks + lk,
                 &As[cur ^ 1][lr * 32 + lk]);
      async_cp16(Ag + (size_t)(lr + 64) * lda + ks + lk,
                 &As[cur ^ 1][(lr + 64) * 32 + lk]);
      async_cp16(Bg + (size_t)lr * ldb + ks + lk,
                 &Bs[cur ^ 1][lr * 32 + lk]);
      __builtin_amdgcn_s_wait_asynccnt(3);  // tile i complete; i+1 in flight
    } else {
      __builtin_amdgcn_s_wait_asynccnt(0);
    }
    __syncthreads();

    V16U a;
    const __bf16* ap = &As[cur][(wave * 16 + mrow) * 32 + half * 8];
    a.h[0] = *(const v8bf*)ap;
    a.h[1] = *(const v8bf*)(ap + 16);
#pragma unroll
    for (int c = 0; c < 4; ++c) {
      V16U b;
      const __bf16* bp = &Bs[cur][(c * 16 + mrow) * 32 + half * 8];
      b.h[0] = *(const v8bf*)bp;
      b.h[1] = *(const v8bf*)(bp + 16);
      acc[c] = __builtin_amdgcn_wmma_f32_16x16x32_bf16(
          false, a.v, false, b.v, (short)0, acc[c], false, false);
    }
    __syncthreads();  // safe to overwrite buf cur^1 next iteration
  }
#else
  // ---- fallback: synchronous staging, single buffer ----
  for (int i = 0; i < steps; ++i) {
    const int ks = i << 5;
    __syncthreads();
    *(v8bf*)&As[0][lr * 32 + lk] =
        *(const v8bf*)(Ag + (size_t)lr * lda + ks + lk);
    *(v8bf*)&As[0][(lr + 64) * 32 + lk] =
        *(const v8bf*)(Ag + (size_t)(lr + 64) * lda + ks + lk);
    *(v8bf*)&Bs[0][lr * 32 + lk] =
        *(const v8bf*)(Bg + (size_t)lr * ldb + ks + lk);
    __syncthreads();

    V16U a;
    const __bf16* ap = &As[0][(wave * 16 + mrow) * 32 + half * 8];
    a.h[0] = *(const v8bf*)ap;
    a.h[1] = *(const v8bf*)(ap + 16);
#pragma unroll
    for (int c = 0; c < 4; ++c) {
      V16U b;
      const __bf16* bp = &Bs[0][(c * 16 + mrow) * 32 + half * 8];
      b.h[0] = *(const v8bf*)bp;
      b.h[1] = *(const v8bf*)(bp + 16);
      acc[c] = __builtin_amdgcn_wmma_f32_16x16x32_bf16(
          false, a.v, false, b.v, (short)0, acc[c], false, false);
    }
  }
#endif

  // C layout: VGPR i -> row = 8*half + i, col = mrow (per 16x16 tile)
  const int gm0 = mBase + wave * 16 + half * 8;
#pragma unroll
  for (int c = 0; c < 4; ++c) {
    const int gn = nBase + c * 16 + mrow;
#pragma unroll
    for (int i = 0; i < 8; ++i) epi(z, gm0 + i, gn, acc[c][i]);
  }
}

// ---------------------------------------------------------------------------
// Epilogues
// ---------------------------------------------------------------------------
struct EpiQKV {
  const float* bias; float* out;
  __device__ void operator()(int, int m, int n, float v) const {
    out[(size_t)m * (3 * CC) + n] = v + bias[n];
  }
};
struct EpiAttn {  // raw attn scores -> d_out attn region [z][m][n]
  float* attn;
  __device__ void operator()(int z, int m, int n, float v) const {
    attn[(size_t)z * NN * NN + (size_t)m * NN + n] = v;
  }
};
struct EpiAV {  // (z=b*H+h, m=token, n=d) -> (B,N,C) scatter, bf16 for proj-A
  __bf16* out;
  __device__ void operator()(int z, int m, int n, float v) const {
    int b = z / HH, h = z - b * HH;
    out[((size_t)b * NN + m) * CC + h * DD + n] = f2bf(v);
  }
};
struct EpiProj {  // + bias + residual x (fp32)
  const float* bias; const float* x; float* src0;
  __device__ void operator()(int, int m, int n, float v) const {
    size_t i = (size_t)m * CC + n;
    src0[i] = v + bias[n] + x[i];
  }
};
struct EpiGelu {  // exact GELU -> bf16 activations
  const float* bias; __bf16* out;
  __device__ void operator()(int, int m, int n, float v) const {
    float t = v + bias[n];
    float g = 0.5f * t * (1.0f + erff(t * 0.70710678118654752440f));
    out[(size_t)m * DFF_ + n] = f2bf(g);
  }
};
struct EpiFC2 {  // + bias + residual src_ln -> d_out src region
  const float* bias; const float* srcln; float* out;
  __device__ void operator()(int, int m, int n, float v) const {
    size_t i = (size_t)m * CC + n;
    out[i] = v + bias[n] + srcln[i];
  }
};

// ---------------------------------------------------------------------------
// Helper kernels
// ---------------------------------------------------------------------------
__global__ void cvt_bf16(const float* __restrict__ s, __bf16* __restrict__ d,
                         int n) {
  int i = blockIdx.x * blockDim.x + threadIdx.x;
  int st = gridDim.x * blockDim.x;
  for (; i < n; i += st) d[i] = f2bf(s[i]);
}

__global__ void zero8(float* p) {
  if (threadIdx.x < 8) p[threadIdx.x] = 0.0f;
}

// Row LayerNorm over C=768, one block (256 thr) per row; bf16 (+opt f32) out
__global__ __launch_bounds__(256) void ln_rows(
    const float* __restrict__ in, const float* __restrict__ w,
    const float* __restrict__ b, __bf16* __restrict__ out_bf,
    float* __restrict__ out_f) {
  const int row = blockIdx.x, tid = threadIdx.x;
  const float* p = in + (size_t)row * CC;
  float v0 = p[tid], v1 = p[tid + 256], v2 = p[tid + 512];
  __shared__ float ss[256], qq[256];
  ss[tid] = v0 + v1 + v2;
  qq[tid] = v0 * v0 + v1 * v1 + v2 * v2;
  __syncthreads();
  for (int o = 128; o > 0; o >>= 1) {
    if (tid < o) { ss[tid] += ss[tid + o]; qq[tid] += qq[tid + o]; }
    __syncthreads();
  }
  float mu = ss[0] * (1.0f / CC);
  float var = qq[0] * (1.0f / CC) - mu * mu;
  float rs = rsqrtf(var + EPS_);
#pragma unroll
  for (int k = 0; k < 3; ++k) {
    int e = tid + k * 256;
    float val = (p[e] - mu) * rs * w[e] + b[e];
    out_bf[(size_t)row * CC + e] = f2bf(val);
    if (out_f) out_f[(size_t)row * CC + e] = val;
  }
}

// One wave per (b,h,n): unit-normalize q,k over D=64, square -> bf16;
// pack v transposed as vt[z][d][n] (bf16) so attn@v's B-operand is K-contig.
__global__ __launch_bounds__(256) void qkv_post(
    const float* __restrict__ qkv, __bf16* __restrict__ qn2,
    __bf16* __restrict__ kn2, __bf16* __restrict__ vt) {
  const int wave = threadIdx.x >> 5, lane = threadIdx.x & 31;
  const int w = blockIdx.x * 8 + wave;  // (b*H+h)*N + n
  const int b = w / (HH * NN);
  const int rem = w - b * HH * NN;
  const int h = rem >> 10;
  const int n = rem & (NN - 1);
  const size_t base = (size_t)(b * NN + n) * (3 * CC);

  float q0 = qkv[base + h * DD + lane];
  float q1 = qkv[base + h * DD + lane + 32];
  float k0 = qkv[base + CC + h * DD + lane];
  float k1 = qkv[base + CC + h * DD + lane + 32];
  float v0 = qkv[base + 2 * CC + h * DD + lane];
  float v1 = qkv[base + 2 * CC + h * DD + lane + 32];

  float sq = q0 * q0 + q1 * q1;
  float sk = k0 * k0 + k1 * k1;
  for (int o = 16; o > 0; o >>= 1) {
    sq += __shfl_xor(sq, o, 32);
    sk += __shfl_xor(sk, o, 32);
  }
  float rq = rsqrtf(sq), rk = rsqrtf(sk);
  q0 *= rq; q1 *= rq; k0 *= rk; k1 *= rk;

  const size_t ob = (size_t)w * DD;  // [z][n][d]
  qn2[ob + lane]      = f2bf(q0 * q0);
  qn2[ob + lane + 32] = f2bf(q1 * q1);
  kn2[ob + lane]      = f2bf(k0 * k0);
  kn2[ob + lane + 32] = f2bf(k1 * k1);
  const size_t vb = (size_t)(b * HH + h) * DD * NN + n;  // [z][d][n]
  vt[vb + (size_t)lane * NN]        = f2bf(v0);
  vt[vb + (size_t)(lane + 32) * NN] = f2bf(v1);
}

// Per-batch sum / sumsq over H*N*N elements (grid.y = batch)
__global__ __launch_bounds__(256) void attn_stats(
    const float* __restrict__ attn, float* __restrict__ stats) {
  const int bidx = blockIdx.y;
  const size_t cnt = (size_t)HH * NN * NN;
  const float* p = attn + (size_t)bidx * cnt;
  float s = 0.f, q = 0.f;
  for (size_t i = (size_t)blockIdx.x * 256 + threadIdx.x; i < cnt;
       i += (size_t)gridDim.x * 256) {
    float v = p[i];
    s += v; q += v * v;
  }
  __shared__ float ss[256], qq[256];
  ss[threadIdx.x] = s; qq[threadIdx.x] = q;
  __syncthreads();
  for (int o = 128; o > 0; o >>= 1) {
    if ((int)threadIdx.x < o) {
      ss[threadIdx.x] += ss[threadIdx.x + o];
      qq[threadIdx.x] += qq[threadIdx.x + o];
    }
    __syncthreads();
  }
  if (threadIdx.x == 0) {
    atomicAdd(stats + bidx * 2, ss[0]);
    atomicAdd(stats + bidx * 2 + 1, qq[0]);
  }
}

// In-place LN apply over (H,N,N) per batch; also emits bf16 shadow (A of attn@v)
__global__ void attn_apply(float* __restrict__ attn,
                           __bf16* __restrict__ attn_bf,
                           const float* __restrict__ w,
                           const float* __restrict__ bb,
                           const float* __restrict__ stats) {
  const size_t per = (size_t)HH * NN * NN;
  const size_t tot = per * BB;
  const float inv = 1.0f / (float)per;
  for (size_t i = (size_t)blockIdx.x * blockDim.x + threadIdx.x; i < tot;
       i += (size_t)gridDim.x * blockDim.x) {
    int b = (int)(i / per);
    size_t j = i - (size_t)b * per;
    float mu = stats[b * 2] * inv;
    float var = stats[b * 2 + 1] * inv - mu * mu;
    float rs = rsqrtf(var + EPS_);
    float r = (attn[i] - mu) * rs * w[j] + bb[j];
    attn[i] = r;
    attn_bf[i] = f2bf(r);
  }
}

// ---------------------------------------------------------------------------
// Launch
// ---------------------------------------------------------------------------
extern "C" void kernel_launch(void* const* d_in, const int* in_sizes, int n_in,
                              void* d_out, int out_size, void* d_ws,
                              size_t ws_size, hipStream_t stream) {
  (void)in_sizes; (void)n_in; (void)out_size; (void)ws_size;
  const float* x      = (const float*)d_in[0];
  const float* ln0_w  = (const float*)d_in[1];
  const float* ln0_b  = (const float*)d_in[2];
  const float* qkv_w  = (const float*)d_in[3];
  const float* qkv_b  = (const float*)d_in[4];
  const float* proj_w = (const float*)d_in[5];
  const float* proj_b = (const float*)d_in[6];
  const float* an_w   = (const float*)d_in[7];
  const float* an_b   = (const float*)d_in[8];
  const float* ln1_w  = (const float*)d_in[9];
  const float* ln1_b  = (const float*)d_in[10];
  const float* fc1_w  = (const float*)d_in[11];
  const float* fc1_b  = (const float*)d_in[12];
  const float* fc2_w  = (const float*)d_in[13];
  const float* fc2_b  = (const float*)d_in[14];

  float* out_src  = (float*)d_out;                   // (B,N,C)
  float* out_attn = out_src + (size_t)BB * NN * CC;  // (B,H,N,N)

  char* cur = (char*)d_ws;
  auto alloc = [&](size_t bytes) -> void* {
    void* p = cur;
    cur += (bytes + 255) & ~(size_t)255;
    return p;
  };
  __bf16* wqkv    = (__bf16*)alloc((size_t)3 * CC * CC * 2);
  __bf16* wproj   = (__bf16*)alloc((size_t)CC * CC * 2);
  __bf16* wfc1    = (__bf16*)alloc((size_t)DFF_ * CC * 2);
  __bf16* wfc2    = (__bf16*)alloc((size_t)CC * DFF_ * 2);
  __bf16* h_bf    = (__bf16*)alloc((size_t)BB * NN * CC * 2);
  float*  qkv_f   = (float*) alloc((size_t)BB * NN * 3 * CC * 4);
  __bf16* qn2     = (__bf16*)alloc((size_t)BB * HH * NN * DD * 2);
  __bf16* kn2     = (__bf16*)alloc((size_t)BB * HH * NN * DD * 2);
  __bf16* vt      = (__bf16*)alloc((size_t)BB * HH * DD * NN * 2);
  __bf16* attn_bf = (__bf16*)alloc((size_t)BB * HH * NN * NN * 2);
  __bf16* avo_bf  = (__bf16*)alloc((size_t)BB * NN * CC * 2);
  float*  src0    = (float*) alloc((size_t)BB * NN * CC * 4);
  float*  srclnf  = (float*) alloc((size_t)BB * NN * CC * 4);
  __bf16* srclnb  = (__bf16*)alloc((size_t)BB * NN * CC * 2);
  __bf16* ff1     = (__bf16*)alloc((size_t)BB * NN * DFF_ * 2);
  float*  stats   = (float*) alloc(64);

  zero8<<<1, 32, 0, stream>>>(stats);
  cvt_bf16<<<1024, 256, 0, stream>>>(qkv_w, wqkv, 3 * CC * CC);
  cvt_bf16<<<512, 256, 0, stream>>>(proj_w, wproj, CC * CC);
  cvt_bf16<<<1024, 256, 0, stream>>>(fc1_w, wfc1, DFF_ * CC);
  cvt_bf16<<<1024, 256, 0, stream>>>(fc2_w, wfc2, CC * DFF_);

  // 1) pre-norm
  ln_rows<<<BB * NN, 256, 0, stream>>>(x, ln0_w, ln0_b, h_bf, nullptr);
  // 2) qkv GEMM: (4096,768) x (2304,768)^T
  gemm128<EpiQKV><<<dim3(3 * CC / 64, BB * NN / 128, 1), 256, 0, stream>>>(
      h_bf, wqkv, CC, CC, CC, 0, 0, EpiQKV{qkv_b, qkv_f});
  // 3) normalize/square q,k ; pack v^T
  qkv_post<<<BB * HH * NN / 8, 256, 0, stream>>>(qkv_f, qn2, kn2, vt);
  // 4) attn = qn2 @ kn2^T per (b,h): 48 x (1024,1024,64) -> d_out attn region
  gemm128<EpiAttn><<<dim3(NN / 64, NN / 128, BB * HH), 256, 0, stream>>>(
      qn2, kn2, DD, DD, DD, (long long)NN * DD, (long long)NN * DD,
      EpiAttn{out_attn});
  // 5) LayerNorm over (H,N,N) per batch: stats then apply (+ bf16 shadow)
  attn_stats<<<dim3(512, BB), 256, 0, stream>>>(out_attn, stats);
  attn_apply<<<2048, 256, 0, stream>>>(out_attn, attn_bf, an_w, an_b, stats);
  // 6) out = attn @ v : 48 x (1024,64,1024) -> bf16 (B,N,C)
  gemm128<EpiAV><<<dim3(1, NN / 128, BB * HH), 256, 0, stream>>>(
      attn_bf, vt, NN, NN, NN, (long long)NN * NN, (long long)DD * NN,
      EpiAV{avo_bf});
  // 7) proj + bias + residual (fp32)
  gemm128<EpiProj><<<dim3(CC / 64, BB * NN / 128, 1), 256, 0, stream>>>(
      avo_bf, wproj, CC, CC, CC, 0, 0, EpiProj{proj_b, x, src0});
  // 8) post-norm
  ln_rows<<<BB * NN, 256, 0, stream>>>(src0, ln1_w, ln1_b, srclnb, srclnf);
  // 9) fc1 + exact GELU
  gemm128<EpiGelu><<<dim3(DFF_ / 64, BB * NN / 128, 1), 256, 0, stream>>>(
      srclnb, wfc1, CC, CC, CC, 0, 0, EpiGelu{fc1_b, ff1});
  // 10) fc2 + bias + residual -> d_out src region
  gemm128<EpiFC2><<<dim3(CC / 64, BB * NN / 128, 1), 256, 0, stream>>>(
      ff1, wfc2, DFF_, DFF_, DFF_, 0, 0, EpiFC2{fc2_b, srclnf, out_src});
}
